// ESSAttn_1236950581626
// MI455X (gfx1250) — matrix operand order repaired and checked by reading, not compiled, not run on hardware
//
#include <hip/hip_runtime.h>
#include <hip/hip_bf16.h>

// ESSAttn for MI455X (gfx1250).
// Primary path: bf16x3 split-GEMM on V_WMMA_F32_16X16X32_BF16 (8x FLOP per
// matrix instruction vs f32 WMMA; fp32 accumulate).
// Fallback path (small workspace): exact-class fp32 V_WMMA_F32_16X16X4_F32.
// b=8, N=4096, C=256. Column-norm of k2 folded into kv row scaling.

typedef float  v2f  __attribute__((ext_vector_type(2)));
typedef float  v8f  __attribute__((ext_vector_type(8)));
typedef __bf16 v8bf __attribute__((ext_vector_type(8)));
typedef __bf16 v16bf __attribute__((ext_vector_type(16)));

#define EPS_DIV  1e-7f
#define EPS_NORM 1e-12f

__device__ __forceinline__ v8f v8f_zero() {
  v8f z = {0.f, 0.f, 0.f, 0.f, 0.f, 0.f, 0.f, 0.f};
  return z;
}

// ---- fp32 WMMA: D = A(16x4) * B(4x16) + C --------------------------------
__device__ __forceinline__ v8f wmma_f32(v2f a, v2f b, v8f c) {
  return __builtin_amdgcn_wmma_f32_16x16x4_f32(false, a, false, b, (short)0, c,
                                               false, false);
}

// ---- bf16 WMMA: D = A(16x32) * B(32x16) + C ------------------------------
__device__ __forceinline__ v8f wmma_bf16(v16bf a, v16bf b, v8f c) {
  return __builtin_amdgcn_wmma_f32_16x16x32_bf16(false, a, false, b, (short)0,
                                                 c, false, false);
}

// A-fragment (16x32 bf16, row-major source): lane holds one contiguous
// 16-element chunk at K = k0+16*half (matches B-fragment order).
__device__ __forceinline__ v16bf load_a_frag(const __bf16* row, int k0,
                                             int half) {
  return *(const v16bf*)(row + k0 + 16 * half);
}
// B-fragment (32x16 bf16): lane n holds one contiguous 16-element chunk at
// K = k0+16*half from the TRANSPOSED operand row BT[n][*].
__device__ __forceinline__ v16bf load_b_frag(const __bf16* rowT, int k0,
                                             int half) {
  return *(const v16bf*)(rowT + k0 + 16 * half);
}

__device__ __forceinline__ void split2(float f, __bf16& hi, __bf16& lo) {
  hi = (__bf16)f;
  lo = (__bf16)(f - (float)hi);
}

__device__ __forceinline__ float wave_sum(float v) {
#pragma unroll
  for (int m = 16; m >= 1; m >>= 1) v += __shfl_xor(v, m, 32);
  return v;
}

// ===========================================================================
// Shared small kernels
// ===========================================================================
__global__ void k_zero(float* __restrict__ p, int n) {
  int i = blockIdx.x * blockDim.x + threadIdx.x;
  if (i < n) p[i] = 0.f;
}

// split fp32 -> bf16 hi/lo, same layout
__global__ void k_split(const float* __restrict__ src, __bf16* __restrict__ hi,
                        __bf16* __restrict__ lo, int n) {
  int i = blockIdx.x * blockDim.x + threadIdx.x;
  if (i < n) split2(src[i], hi[i], lo[i]);
}

// W[k][n] (KxN fp32) -> WT_hi/lo[n][k] (bf16)
__global__ void k_split_wT(const float* __restrict__ W, __bf16* __restrict__ hi,
                           __bf16* __restrict__ lo, int K, int N) {
  int i = blockIdx.x * blockDim.x + threadIdx.x;  // output index [n][k]
  if (i >= K * N) return;
  int n = i / K, k = i % K;
  split2(W[k * N + n], hi[i], lo[i]);
}

// ---------------------------------------------------------------------------
// Centering/normalization (one wave per token, 64 tokens/block).
//   q-slice <- l2_normalize(q2/(q2s+eps)) over C
//     BF16Q=1: packed bf16 hi[0..255] lo[256..511] in place of q fp32 slice
//     BF16Q=0: fp32 in place
//   k-slice <- z = k2/(k2s+eps) fp32 (column norm deferred to kv scaling)
//   colss[b,c] += sum_n z^2
// ---------------------------------------------------------------------------
template <bool BF16Q>
__global__ __launch_bounds__(256) void k_center(float* __restrict__ qkv,
                                                float* __restrict__ colss) {
  __shared__ float lds_css[256];
  const int tid = threadIdx.x;
  const int wave = tid >> 5, lane = tid & 31;
  lds_css[tid] = 0.f;
  __syncthreads();

  const int tok0 = blockIdx.x * 64 + wave * 8;
  const int b = (blockIdx.x * 64) >> 12;
  float zacc[8] = {0.f, 0.f, 0.f, 0.f, 0.f, 0.f, 0.f, 0.f};

  for (int i = 0; i < 8; ++i) {
    float* row = qkv + (size_t)(tok0 + i) * 768;
    float qv[8], kw[8];
    float sq = 0.f, sk = 0.f;
#pragma unroll
    for (int j = 0; j < 8; ++j) {
      qv[j] = row[lane + 32 * j];
      kw[j] = row[256 + lane + 32 * j];
      sq += qv[j];
      sk += kw[j];
    }
    const float mq = wave_sum(sq) * (1.f / 256.f);
    const float mk = wave_sum(sk) * (1.f / 256.f);
    float q2s = 0.f, k2s = 0.f;
#pragma unroll
    for (int j = 0; j < 8; ++j) {
      const float qc = qv[j] - mq; qv[j] = qc * qc; q2s += qv[j];
      const float kc = kw[j] - mk; kw[j] = kc * kc; k2s += kw[j];
    }
    q2s = wave_sum(q2s);
    k2s = wave_sum(k2s);
    const float qinv = 1.f / (q2s + EPS_DIV);
    const float kinv = 1.f / (k2s + EPS_DIV);
    float ss = 0.f;
#pragma unroll
    for (int j = 0; j < 8; ++j) { qv[j] *= qinv; ss += qv[j] * qv[j]; }
    ss = wave_sum(ss);
    const float qn = 1.f / fmaxf(sqrtf(ss), EPS_NORM);
#pragma unroll
    for (int j = 0; j < 8; ++j) {
      const int c = lane + 32 * j;
      const float qfin = qv[j] * qn;
      if (BF16Q) {
        __bf16* qrow = (__bf16*)row;  // 512 bf16 slots over bytes [0,1024)
        split2(qfin, qrow[c], qrow[256 + c]);
      } else {
        row[c] = qfin;
      }
      const float z = kw[j] * kinv;
      row[256 + c] = z;
      zacc[j] += z * z;
    }
  }
#pragma unroll
  for (int j = 0; j < 8; ++j) atomicAdd(&lds_css[lane + 32 * j], zacc[j]);
  __syncthreads();
  atomicAdd(&colss[b * 256 + tid], lds_css[tid]);
}

// ===========================================================================
// bf16x3 path
// ===========================================================================

// qkv[32768,768] = x @ W_qkv + b_qkv; one wave -> 16x64 tile; bf16x3.
__global__ __launch_bounds__(128) void k_gemm_qkv_bf(
    const __bf16* __restrict__ xhi, const __bf16* __restrict__ xlo,
    const __bf16* __restrict__ wthi, const __bf16* __restrict__ wtlo,
    const float* __restrict__ bq, float* __restrict__ qkv) {
  const int wave = threadIdx.x >> 5, lane = threadIdx.x & 31;
  const int half = lane >> 4, l16 = lane & 15;
  const int tile = blockIdx.x * 4 + wave;
  const int m0 = (tile / 12) * 16;
  const int n0 = (tile % 12) * 64;
  const __bf16* Ah = xhi + (size_t)(m0 + l16) * 256;
  const __bf16* Al = xlo + (size_t)(m0 + l16) * 256;
  v8f acc[4];
#pragma unroll
  for (int j = 0; j < 4; ++j) acc[j] = v8f_zero();

#pragma unroll
  for (int k0 = 0; k0 < 256; k0 += 32) {
    const v16bf ah = load_a_frag(Ah, k0, half);
    const v16bf al = load_a_frag(Al, k0, half);
#pragma unroll
    for (int j = 0; j < 4; ++j) {
      const size_t nrow = (size_t)(n0 + 16 * j + l16) * 256;
      const v16bf bh = load_b_frag(wthi + nrow, k0, half);
      const v16bf bl = load_b_frag(wtlo + nrow, k0, half);
      acc[j] = wmma_bf16(ah, bh, acc[j]);
      acc[j] = wmma_bf16(ah, bl, acc[j]);
      acc[j] = wmma_bf16(al, bh, acc[j]);
    }
  }
#pragma unroll
  for (int j = 0; j < 4; ++j) {
    const int col = n0 + 16 * j + l16;
    const float bias = bq[col];
#pragma unroll
    for (int r = 0; r < 8; ++r)
      qkv[(size_t)(m0 + r + 8 * half) * 768 + col] = acc[j][r] + bias;
  }
}

// Tiled transpose+split of k2c and v slices: qkv[n][c] -> T_hi/lo[b][c][n].
__global__ __launch_bounds__(256) void k_trans_split(
    const float* __restrict__ qkv, __bf16* __restrict__ k2thi,
    __bf16* __restrict__ k2tlo, __bf16* __restrict__ vthi,
    __bf16* __restrict__ vtlo) {
  __shared__ float tk[32][33];
  __shared__ float tv[32][33];
  const int tid = threadIdx.x;
  const int tx = tid & 31, ty = tid >> 5;  // ty in 0..7
  const int bidx = blockIdx.x;             // 8 * 128 * 8 = 8192 blocks
  const int b = bidx >> 10;
  const int t = bidx & 1023;
  const int n0 = (t >> 3) * 32;
  const int c0 = (t & 7) * 32;

#pragma unroll
  for (int i = 0; i < 4; ++i) {
    const int nr = ty + 8 * i;
    const float* r = qkv + (size_t)(b * 4096 + n0 + nr) * 768;
    tk[nr][tx] = r[256 + c0 + tx];
    tv[nr][tx] = r[512 + c0 + tx];
  }
  __syncthreads();
#pragma unroll
  for (int i = 0; i < 4; ++i) {
    const int cr = ty + 8 * i;  // local c row of output
    const size_t o = ((size_t)b * 256 + c0 + cr) * 4096 + n0 + tx;
    split2(tk[tx][cr], k2thi[o], k2tlo[o]);
    split2(tv[tx][cr], vthi[o], vtlo[o]);
  }
}

// kv_raw[b][c][d] = sum_n k2c[n][c] * v[n][d]; one wave -> 16x64; K=4096.
__global__ __launch_bounds__(128) void k_gemm_kv_bf(
    const __bf16* __restrict__ k2thi, const __bf16* __restrict__ k2tlo,
    const __bf16* __restrict__ vthi, const __bf16* __restrict__ vtlo,
    float* __restrict__ kvm) {
  const int wave = threadIdx.x >> 5, lane = threadIdx.x & 31;
  const int half = lane >> 4, l16 = lane & 15;
  const int tile = blockIdx.x * 4 + wave;  // 0..511
  const int b = tile >> 6;
  const int t = tile & 63;
  const int c0 = (t >> 2) * 16;
  const int d0 = (t & 3) * 64;
  const __bf16* Ah = k2thi + ((size_t)b * 256 + c0 + l16) * 4096;
  const __bf16* Al = k2tlo + ((size_t)b * 256 + c0 + l16) * 4096;
  const __bf16* Bh = vthi + ((size_t)b * 256 + d0 + l16) * 4096;
  const __bf16* Bl = vtlo + ((size_t)b * 256 + d0 + l16) * 4096;
  v8f acc[4];
#pragma unroll
  for (int j = 0; j < 4; ++j) acc[j] = v8f_zero();

#pragma unroll 2
  for (int k0 = 0; k0 < 4096; k0 += 32) {
    const v16bf ah = load_a_frag(Ah, k0, half);
    const v16bf al = load_a_frag(Al, k0, half);
#pragma unroll
    for (int j = 0; j < 4; ++j) {
      const v16bf bh = load_b_frag(Bh + (size_t)16 * j * 4096, k0, half);
      const v16bf bl = load_b_frag(Bl + (size_t)16 * j * 4096, k0, half);
      acc[j] = wmma_bf16(ah, bh, acc[j]);
      acc[j] = wmma_bf16(ah, bl, acc[j]);
      acc[j] = wmma_bf16(al, bh, acc[j]);
    }
  }
  float* D = kvm + (size_t)b * 65536;
#pragma unroll
  for (int j = 0; j < 4; ++j)
#pragma unroll
    for (int r = 0; r < 8; ++r)
      D[(size_t)(c0 + r + 8 * half) * 256 + d0 + 16 * j + l16] = acc[j][r];
}

// kvT_hi/lo[b][d][c] = kv[b][c][d] * (1/64) / max(sqrt(colss[b,c]), eps)
__global__ void k_scale_kvT(const float* __restrict__ kvm,
                            const float* __restrict__ colss,
                            __bf16* __restrict__ kvthi,
                            __bf16* __restrict__ kvtlo) {
  const int idx = blockIdx.x * 256 + threadIdx.x;
  const int b = idx >> 16, c = (idx >> 8) & 255, d = idx & 255;
  const float s = colss[b * 256 + c];
  const float val = kvm[idx] * (0.015625f / fmaxf(sqrtf(s), EPS_NORM));
  const size_t o = ((size_t)(b * 256 + d)) * 256 + c;
  split2(val, kvthi[o], kvtlo[o]);
}

// Fused: t2 = q2n @ kv; attn = v + t2 (split into LDS once);
//        out = attn @ W_out + b_out. 16 waves/block, bf16x3 both phases.
__global__ __launch_bounds__(512) void k_out_bf(
    const float* __restrict__ qkv, const __bf16* __restrict__ kvthi,
    const __bf16* __restrict__ kvtlo, const __bf16* __restrict__ wothi,
    const __bf16* __restrict__ wotlo, const float* __restrict__ bo,
    float* __restrict__ out) {
  __shared__ __align__(32) __bf16 lath[16][272];
  __shared__ __align__(32) __bf16 latl[16][272];
  const int tid = threadIdx.x;
  const int wave = tid >> 5, lane = tid & 31;
  const int half = lane >> 4, l16 = lane & 15;
  const int m0 = blockIdx.x * 16;
  const int b = m0 >> 12;
  const int col0 = wave * 16;

  // Phase 1: t2 tile; A = packed q2n hi/lo in q-slice, B = kvT rows.
  const __bf16* qrow = (const __bf16*)(qkv + (size_t)(m0 + l16) * 768);
  const __bf16* kvth = kvthi + ((size_t)b * 256 + col0 + l16) * 256;
  const __bf16* kvtl = kvtlo + ((size_t)b * 256 + col0 + l16) * 256;
  v8f acc = v8f_zero();
#pragma unroll
  for (int k0 = 0; k0 < 256; k0 += 32) {
    const v16bf ah = load_a_frag(qrow, k0, half);
    const v16bf al = load_a_frag(qrow + 256, k0, half);
    const v16bf bh = load_b_frag(kvth, k0, half);
    const v16bf bl = load_b_frag(kvtl, k0, half);
    acc = wmma_bf16(ah, bh, acc);
    acc = wmma_bf16(ah, bl, acc);
    acc = wmma_bf16(al, bh, acc);
  }
#pragma unroll
  for (int r = 0; r < 8; ++r) {
    const int rl = r + 8 * half;
    const float attn =
        acc[r] + qkv[(size_t)(m0 + rl) * 768 + 512 + col0 + l16];
    __bf16 h, l;
    split2(attn, h, l);
    lath[rl][col0 + l16] = h;
    latl[rl][col0 + l16] = l;
  }
  __syncthreads();

  // Phase 2: out tile = attn @ W_out + b_out
  v8f acc2 = v8f_zero();
  const __bf16* woh = wothi + (size_t)(col0 + l16) * 256;
  const __bf16* wol = wotlo + (size_t)(col0 + l16) * 256;
#pragma unroll
  for (int k0 = 0; k0 < 256; k0 += 32) {
    const v16bf ah = load_a_frag(&lath[l16][0], k0, half);
    const v16bf al = load_a_frag(&latl[l16][0], k0, half);
    const v16bf bh = load_b_frag(woh, k0, half);
    const v16bf bl = load_b_frag(wol, k0, half);
    acc2 = wmma_bf16(ah, bh, acc2);
    acc2 = wmma_bf16(ah, bl, acc2);
    acc2 = wmma_bf16(al, bh, acc2);
  }
  const float bias = bo[col0 + l16];
#pragma unroll
  for (int r = 0; r < 8; ++r)
    out[(size_t)(m0 + r + 8 * half) * 256 + col0 + l16] = acc2[r] + bias;
}

// ===========================================================================
// fp32 WMMA fallback path (small workspace)
// ===========================================================================
__global__ __launch_bounds__(128) void k_gemm_qkv_f32(
    const float* __restrict__ x, const float* __restrict__ Wq,
    const float* __restrict__ bq, float* __restrict__ qkv) {
  const int wave = threadIdx.x >> 5, lane = threadIdx.x & 31;
  const int half = lane >> 4, l16 = lane & 15;
  const int tile = blockIdx.x * 4 + wave;
  const int m0 = (tile / 12) * 16;
  const int n0 = (tile % 12) * 64;
  const float* A = x + (size_t)(m0 + l16) * 256;
  v8f acc[4];
#pragma unroll
  for (int j = 0; j < 4; ++j) acc[j] = v8f_zero();
#pragma unroll 4
  for (int k = 0; k < 256; k += 4) {
    const int ka = k + 2 * half;
    v2f a; a.x = A[ka]; a.y = A[ka + 1];
    const float* Bp = Wq + (size_t)ka * 768 + n0 + l16;
#pragma unroll
    for (int j = 0; j < 4; ++j) {
      v2f bb; bb.x = Bp[16 * j]; bb.y = Bp[768 + 16 * j];
      acc[j] = wmma_f32(a, bb, acc[j]);
    }
  }
#pragma unroll
  for (int j = 0; j < 4; ++j) {
    const int col = n0 + 16 * j + l16;
    const float bias = bq[col];
#pragma unroll
    for (int r = 0; r < 8; ++r)
      qkv[(size_t)(m0 + r + 8 * half) * 768 + col] = acc[j][r] + bias;
  }
}

__global__ __launch_bounds__(128) void k_gemm_kv_f32(
    const float* __restrict__ qkv, float* __restrict__ kvm) {
  const int wave = threadIdx.x >> 5, lane = threadIdx.x & 31;
  const int half = lane >> 4, l16 = lane & 15;
  const int tile = blockIdx.x * 4 + wave;
  const int b = tile >> 6;
  const int t = tile & 63;
  const int c0 = (t >> 2) * 16;
  const int d0 = (t & 3) * 64;
  const float* kbase = qkv + (size_t)b * 4096 * 768 + 256;
  const float* vbase = qkv + (size_t)b * 4096 * 768 + 512;
  v8f acc[4];
#pragma unroll
  for (int j = 0; j < 4; ++j) acc[j] = v8f_zero();
#pragma unroll 4
  for (int n = 0; n < 4096; n += 4) {
    const int na = n + 2 * half;
    v2f a;
    a.x = kbase[(size_t)na * 768 + c0 + l16];
    a.y = kbase[(size_t)(na + 1) * 768 + c0 + l16];
    const float* Bp = vbase + (size_t)na * 768 + d0 + l16;
#pragma unroll
    for (int j = 0; j < 4; ++j) {
      v2f bb; bb.x = Bp[16 * j]; bb.y = Bp[768 + 16 * j];
      acc[j] = wmma_f32(a, bb, acc[j]);
    }
  }
  float* D = kvm + (size_t)b * 65536;
#pragma unroll
  for (int j = 0; j < 4; ++j)
#pragma unroll
    for (int r = 0; r < 8; ++r)
      D[(size_t)(c0 + r + 8 * half) * 256 + d0 + 16 * j + l16] = acc[j][r];
}

__global__ void k_scale_kv_f32(float* __restrict__ kvm,
                               const float* __restrict__ colss) {
  const int idx = blockIdx.x * 256 + threadIdx.x;
  const int b = idx >> 16;
  const int c = (idx >> 8) & 255;
  const float s = colss[b * 256 + c];
  kvm[idx] *= 0.015625f / fmaxf(sqrtf(s), EPS_NORM);
}

__global__ __launch_bounds__(512) void k_out_f32(
    const float* __restrict__ qkv, const float* __restrict__ kvm,
    const float* __restrict__ Wo, const float* __restrict__ bo,
    float* __restrict__ out) {
  __shared__ float lat[16][257];
  const int tid = threadIdx.x;
  const int wave = tid >> 5, lane = tid & 31;
  const int half = lane >> 4, l16 = lane & 15;
  const int m0 = blockIdx.x * 16;
  const int b = m0 >> 12;
  const int col0 = wave * 16;
  const float* A = qkv + (size_t)(m0 + l16) * 768;
  const float* KV = kvm + (size_t)b * 65536;
  v8f acc = v8f_zero();
#pragma unroll 4
  for (int c = 0; c < 256; c += 4) {
    const int ca = c + 2 * half;
    v2f a; a.x = A[ca]; a.y = A[ca + 1];
    v2f bb;
    bb.x = KV[(size_t)ca * 256 + col0 + l16];
    bb.y = KV[(size_t)(ca + 1) * 256 + col0 + l16];
    acc = wmma_f32(a, bb, acc);
  }
#pragma unroll
  for (int r = 0; r < 8; ++r) {
    const int rl = r + 8 * half;
    const float vv = qkv[(size_t)(m0 + rl) * 768 + 512 + col0 + l16];
    lat[rl][col0 + l16] = acc[r] + vv;
  }
  __syncthreads();
  v8f acc2 = v8f_zero();
#pragma unroll 4
  for (int d = 0; d < 256; d += 4) {
    const int da = d + 2 * half;
    v2f a; a.x = lat[l16][da]; a.y = lat[l16][da + 1];
    v2f bb;
    bb.x = Wo[(size_t)da * 256 + col0 + l16];
    bb.y = Wo[(size_t)(da + 1) * 256 + col0 + l16];
    acc2 = wmma_f32(a, bb, acc2);
  }
  const float bias = bo[col0 + l16];
#pragma unroll
  for (int r = 0; r < 8; ++r)
    out[(size_t)(m0 + r + 8 * half) * 256 + col0 + l16] = acc2[r] + bias;
}

// ===========================================================================
extern "C" void kernel_launch(void* const* d_in, const int* in_sizes, int n_in,
                              void* d_out, int out_size, void* d_ws,
                              size_t ws_size, hipStream_t stream) {
  (void)in_sizes; (void)n_in; (void)out_size;
  const float* x  = (const float*)d_in[0];
  const float* Wq = (const float*)d_in[1];
  const float* bq = (const float*)d_in[2];
  const float* Wo = (const float*)d_in[3];
  const float* bo = (const float*)d_in[4];
  float* out = (float*)d_out;

  char* w = (char*)d_ws;
  constexpr size_t QKV_B = (size_t)32768 * 768 * 4;  // 100,663,296

  // bf16 path workspace layout
  size_t o = QKV_B;
  float*  qkv   = (float*)w;
  __bf16* xhi   = (__bf16*)(w + o); o += (size_t)32768 * 256 * 2;  // ->k2cT_hi
  __bf16* xlo   = (__bf16*)(w + o); o += (size_t)32768 * 256 * 2;  // ->k2cT_lo
  __bf16* vthi  = (__bf16*)(w + o); o += (size_t)32768 * 256 * 2;
  __bf16* vtlo  = (__bf16*)(w + o); o += (size_t)32768 * 256 * 2;
  __bf16* wqthi = (__bf16*)(w + o); o += (size_t)768 * 256 * 2;
  __bf16* wqtlo = (__bf16*)(w + o); o += (size_t)768 * 256 * 2;
  __bf16* wothi = (__bf16*)(w + o); o += (size_t)256 * 256 * 2;
  __bf16* wotlo = (__bf16*)(w + o); o += (size_t)256 * 256 * 2;
  float*  kvm   = (float*)(w + o);  o += (size_t)8 * 256 * 256 * 4;
  __bf16* kvthi = (__bf16*)(w + o); o += (size_t)8 * 256 * 256 * 2;
  __bf16* kvtlo = (__bf16*)(w + o); o += (size_t)8 * 256 * 256 * 2;
  float*  colss = (float*)(w + o);  o += 8192;
  const size_t NEED_BF16 = o;                       // ~173 MB
  const size_t NEED_F32 = QKV_B + 2097152 + 8192;   // ~103 MB

  if (ws_size >= NEED_BF16) {
    // bf16x3 split-GEMM path (8x FLOP/instruction on the matrix pipes)
    __bf16* k2thi = xhi;  // x splits dead after k_gemm_qkv_bf -> reuse
    __bf16* k2tlo = xlo;
    k_zero<<<8, 256, 0, stream>>>(colss, 2048);
    k_split<<<32768, 256, 0, stream>>>(x, xhi, xlo, 32768 * 256);
    k_split_wT<<<768, 256, 0, stream>>>(Wq, wqthi, wqtlo, 256, 768);
    k_split_wT<<<256, 256, 0, stream>>>(Wo, wothi, wotlo, 256, 256);
    k_gemm_qkv_bf<<<6144, 128, 0, stream>>>(xhi, xlo, wqthi, wqtlo, bq, qkv);
    k_center<true><<<512, 256, 0, stream>>>(qkv, colss);
    k_trans_split<<<8192, 256, 0, stream>>>(qkv, k2thi, k2tlo, vthi, vtlo);
    k_gemm_kv_bf<<<128, 128, 0, stream>>>(k2thi, k2tlo, vthi, vtlo, kvm);
    k_scale_kvT<<<2048, 256, 0, stream>>>(kvm, colss, kvthi, kvtlo);
    k_out_bf<<<2048, 512, 0, stream>>>(qkv, kvthi, kvtlo, wothi, wotlo, bo,
                                       out);
  } else {
    // fp32 WMMA fallback (exact precision class, smaller workspace)
    (void)NEED_F32;
    float* kvm2 = (float*)(w + QKV_B);
    float* colss2 = (float*)(w + QKV_B + 2097152);
    k_zero<<<8, 256, 0, stream>>>(colss2, 2048);
    k_gemm_qkv_f32<<<6144, 128, 0, stream>>>(x, Wq, bq, qkv);
    k_center<false><<<512, 256, 0, stream>>>(qkv, colss2);
    k_gemm_kv_f32<<<128, 128, 0, stream>>>(qkv, kvm2);
    k_scale_kv_f32<<<2048, 256, 0, stream>>>(kvm2, colss2);
    k_out_f32<<<2048, 512, 0, stream>>>(qkv, kvm2, Wo, bo, out);
  }
}
